// EncoderLayer2_75969381532482
// MI455X (gfx1250) — compile-verified, matrix operand order checked by reading, not run on hardware
//
#include <hip/hip_runtime.h>
#include <math.h>

#define BLOCK 256
#define WAVES 8

typedef __attribute__((ext_vector_type(16))) _Float16 v16h;
typedef __attribute__((ext_vector_type(8)))  float    v8f;

// fast activations: v_rcp_f32 / v_exp_f32 / v_log_f32 (no IEEE div fixup chain)
__device__ __forceinline__ float silu_f(float v) {
  return v * __builtin_amdgcn_rcpf(1.f + __expf(-v));
}
__device__ __forceinline__ float softplus_f(float v) {
  return v > 20.f ? v : __logf(1.f + __expf(v));
}

// ---- WMMA fragment helpers (CDNA5 16-bit 16x32 A layout, ISA 7.12.2) ----
// A: lanes 0-15 hold row m, K in {0..7, 16..23}; lanes 16-31 hold row m, K in {8..15, 24..31}
__device__ __forceinline__ v16h frag_a(const float* __restrict__ src, int rs, int cs,
                                       int mbase, int kbase, int lane) {
  int m  = mbase + (lane & 15);
  int kh = (lane >> 4) << 3;
  v16h f;
#pragma unroll
  for (int h = 0; h < 16; ++h) {
    int k = kbase + ((h < 8) ? h : (h + 8)) + kh;
    f[h] = (_Float16)src[m * rs + k * cs];
  }
  return f;
}

// B[k,n] = W[n*ldk + k] (weight is row-major N x K); lane = column n, same K split as A
__device__ __forceinline__ v16h frag_bT(const float* __restrict__ W, int ldk,
                                        int nbase, int kbase, int lane, int nmax) {
  int n  = nbase + (lane & 15);
  int kh = (lane >> 4) << 3;
  bool ok = n < nmax;
  v16h f;
#pragma unroll
  for (int h = 0; h < 16; ++h) {
    int k = kbase + ((h < 8) ? h : (h + 8)) + kh;
    f[h] = ok ? (_Float16)W[n * ldk + k] : (_Float16)0.f;
  }
  return f;
}

// D/C: VGPR r -> row r (lanes 0-15) or row r+8 (lanes 16-31); col = lane&15
__device__ __forceinline__ void store_tile(float* __restrict__ C, int ldc,
                                           int mbase, int nbase, v8f d, int lane) {
  int col  = nbase + (lane & 15);
  int row0 = mbase + ((lane >> 4) << 3);
#pragma unroll
  for (int r = 0; r < 8; ++r)
    C[(row0 + r) * ldc + col] = d[r];
}

// C(M x Npad) = A(M x K, strides rs/cs) * W^T, W row-major (nmaxW x K). 8 waves/block.
__device__ __forceinline__ void gemm_wmma(const float* __restrict__ A, int ars, int acs,
                                          const float* __restrict__ W, int ldk, int nmaxW,
                                          float* __restrict__ C, int ldc,
                                          int M, int Npad, int K, int tid) {
  int wave = tid >> 5, lane = tid & 31;
  int ntl = Npad >> 4;
  int T = (M >> 4) * ntl;
  for (int t = wave; t < T; t += WAVES) {   // wave-uniform: EXEC stays all-ones for WMMA
    int mt = t / ntl;
    int mb = mt << 4;
    int nb = (t - mt * ntl) << 4;
    v8f acc = {};
#pragma unroll 2
    for (int kb = 0; kb < K; kb += 32) {
      v16h a = frag_a(A, ars, acs, mb, kb, lane);
      v16h b = frag_bT(W, ldk, nb, kb, lane, nmaxW);
      acc = __builtin_amdgcn_wmma_f32_16x16x32_f16(false, a, false, b, (short)0, acc,
                                                   false, false);
    }
    store_tile(C, ldc, mb, nb, acc, lane);
  }
}

// ===================== Branch 1: Mamba over (B, L=64, DM=128) =====================
__global__ __launch_bounds__(BLOCK)
void mamba1_kernel(const float* __restrict__ x,
                   const float* __restrict__ in_w,  const float* __restrict__ cw,
                   const float* __restrict__ cb,    const float* __restrict__ xp_w,
                   const float* __restrict__ dt_w,  const float* __restrict__ dt_b,
                   const float* __restrict__ A_log, const float* __restrict__ Dv,
                   const float* __restrict__ out_w, const float* __restrict__ nw,
                   const float* __restrict__ nb,    float* __restrict__ out) {
  extern __shared__ float sm[];
  const int L = 64, DM = 128;
  float* xz  = sm;               // 64 x 256 : [u | z]
  float* su  = xz + L * 2 * DM;  // 64 x 128 : u (later gated y)
  float* xd  = su + L * DM;      // 64 x 48  : x_dbl (40 valid, padded)
  float* sdt = xd + L * 48;      // 64 x 128 : dt (later y)
  const int b = blockIdx.x, tid = threadIdx.x;
  const float* xb = x + b * L * DM;

  // in-proj: xz = x @ in_w^T   (64x128 * 128x256)
  gemm_wmma(xb, DM, 1, in_w, DM, 2 * DM, xz, 2 * DM, L, 2 * DM, DM, tid);
  __syncthreads();

  // causal depthwise conv (K=2) + SiLU
  for (int i = tid; i < L * DM; i += BLOCK) {
    int l = i >> 7, d = i & 127;
    float cur  = xz[l * 256 + d];
    float prev = l ? xz[(l - 1) * 256 + d] : 0.f;
    su[i] = silu_f(prev * cw[d * 2] + cur * cw[d * 2 + 1] + cb[d]);
  }
  __syncthreads();

  // x-proj: x_dbl = u @ xp_w^T  (N=40 padded to 48)
  gemm_wmma(su, DM, 1, xp_w, DM, 40, xd, 48, L, 48, DM, tid);
  __syncthreads();

  // dt = softplus(x_dbl[:, :8] @ dt_w^T + dt_b)
  for (int i = tid; i < L * DM; i += BLOCK) {
    int l = i >> 7, d = i & 127;
    float acc = dt_b[d];
#pragma unroll
    for (int r = 0; r < 8; ++r) acc += xd[l * 48 + r] * dt_w[d * 8 + r];
    sdt[i] = softplus_f(acc);
  }
  __syncthreads();

  // selective scan: one thread per channel, 16 states in registers
  if (tid < DM) {
    int d = tid;
    float Ar[16], h[16];
#pragma unroll
    for (int n = 0; n < 16; ++n) { Ar[n] = -__expf(A_log[d * 16 + n]); h[n] = 0.f; }
    float Dd = Dv[d];
    for (int l = 0; l < L; ++l) {
      float dl = sdt[l * DM + d];
      float ul = su[l * DM + d];
      float acc = 0.f;
#pragma unroll
      for (int n = 0; n < 16; ++n) {
        h[n] = h[n] * __expf(dl * Ar[n]) + dl * xd[l * 48 + 8 + n] * ul;
        acc += h[n] * xd[l * 48 + 24 + n];
      }
      sdt[l * DM + d] = acc + ul * Dd;   // y overwrites dt in place
    }
  }
  __syncthreads();

  // gate: g = y * silu(z)
  for (int i = tid; i < L * DM; i += BLOCK) {
    int l = i >> 7, d = i & 127;
    su[i] = sdt[i] * silu_f(xz[l * 256 + DM + d]);
  }
  __syncthreads();

  // out-proj: a = g @ out_w^T (reuse xz as 64x128 buffer)
  gemm_wmma(su, DM, 1, out_w, DM, DM, xz, DM, L, DM, DM, tid);
  __syncthreads();

  // LayerNorm over DM, write to d_out
  if (tid < L) {
    int l = tid;
    float s = 0.f, s2 = 0.f;
    for (int d = 0; d < DM; ++d) { float v = xz[l * DM + d]; s += v; s2 += v * v; }
    float m = s * (1.f / DM);
    float inv = rsqrtf(s2 * (1.f / DM) - m * m + 1e-5f);
    for (int d = 0; d < DM; ++d)
      out[b * L * DM + l * DM + d] = (xz[l * DM + d] - m) * inv * nw[d] + nb[d];
  }
}

// ============ Branch 2: Mamba over x^T (B, 128, 64), result added transposed ============
__global__ __launch_bounds__(BLOCK)
void mamba2_kernel(const float* __restrict__ x,
                   const float* __restrict__ in_w,  const float* __restrict__ cw,
                   const float* __restrict__ cb,    const float* __restrict__ xp_w,
                   const float* __restrict__ dt_w,  const float* __restrict__ dt_b,
                   const float* __restrict__ A_log, const float* __restrict__ Dv,
                   const float* __restrict__ out_w, const float* __restrict__ nw,
                   const float* __restrict__ nb,    float* __restrict__ out) {
  extern __shared__ float sm[];
  const int L2 = 128, D2 = 64;
  float* xz  = sm;                // 128 x 128
  float* su  = xz + L2 * 2 * D2;  // 128 x 64
  float* xd  = su + L2 * D2;      // 128 x 48 (36 valid)
  float* sdt = xd + L2 * 48;      // 128 x 64
  const int b = blockIdx.x, tid = threadIdx.x;
  const float* xb = x + b * 64 * 128;  // A(m,k) = xb[k*128 + m] (transposed view)

  gemm_wmma(xb, 1, 128, in_w, D2, 2 * D2, xz, 2 * D2, L2, 2 * D2, D2, tid);
  __syncthreads();

  for (int i = tid; i < L2 * D2; i += BLOCK) {
    int l = i >> 6, d = i & 63;
    float cur  = xz[l * 128 + d];
    float prev = l ? xz[(l - 1) * 128 + d] : 0.f;
    su[i] = silu_f(prev * cw[d * 2] + cur * cw[d * 2 + 1] + cb[d]);
  }
  __syncthreads();

  gemm_wmma(su, D2, 1, xp_w, D2, 36, xd, 48, L2, 48, D2, tid);
  __syncthreads();

  for (int i = tid; i < L2 * D2; i += BLOCK) {
    int l = i >> 6, d = i & 63;
    float acc = dt_b[d];
#pragma unroll
    for (int r = 0; r < 4; ++r) acc += xd[l * 48 + r] * dt_w[d * 4 + r];
    sdt[i] = softplus_f(acc);
  }
  __syncthreads();

  if (tid < D2) {
    int d = tid;
    float Ar[16], h[16];
#pragma unroll
    for (int n = 0; n < 16; ++n) { Ar[n] = -__expf(A_log[d * 16 + n]); h[n] = 0.f; }
    float Dd = Dv[d];
    for (int l = 0; l < L2; ++l) {
      float dl = sdt[l * D2 + d];
      float ul = su[l * D2 + d];
      float acc = 0.f;
#pragma unroll
      for (int n = 0; n < 16; ++n) {
        h[n] = h[n] * __expf(dl * Ar[n]) + dl * xd[l * 48 + 4 + n] * ul;
        acc += h[n] * xd[l * 48 + 20 + n];
      }
      sdt[l * D2 + d] = acc + ul * Dd;
    }
  }
  __syncthreads();

  for (int i = tid; i < L2 * D2; i += BLOCK) {
    int l = i >> 6, d = i & 63;
    su[i] = sdt[i] * silu_f(xz[l * 128 + D2 + d]);
  }
  __syncthreads();

  gemm_wmma(su, D2, 1, out_w, D2, D2, xz, D2, L2, D2, D2, tid);
  __syncthreads();

  // LayerNorm over D2, then out[b, j, i] += val (transpose-add)
  if (tid < L2) {
    int i = tid;
    float s = 0.f, s2 = 0.f;
    for (int j = 0; j < D2; ++j) { float v = xz[i * D2 + j]; s += v; s2 += v * v; }
    float m = s * (1.f / D2);
    float inv = rsqrtf(s2 * (1.f / D2) - m * m + 1e-5f);
    for (int j = 0; j < D2; ++j) {
      float v = (xz[i * D2 + j] - m) * inv * nw[j] + nb[j];
      int o = b * 8192 + j * 128 + i;
      out[o] = out[o] + v;
    }
  }
}

extern "C" void kernel_launch(void* const* d_in, const int* in_sizes, int n_in,
                              void* d_out, int out_size, void* d_ws, size_t ws_size,
                              hipStream_t stream) {
  (void)in_sizes; (void)n_in; (void)d_ws; (void)ws_size; (void)out_size;
  const float* x = (const float*)d_in[0];
  float* out = (float*)d_out;

  size_t lds1 = (size_t)(64 * 256 + 64 * 128 + 64 * 48 + 64 * 128) * sizeof(float);   // ~140 KB
  size_t lds2 = (size_t)(128 * 128 + 128 * 64 + 128 * 48 + 128 * 64) * sizeof(float); // ~152 KB

  mamba1_kernel<<<256, BLOCK, lds1, stream>>>(
      x,
      (const float*)d_in[1],  (const float*)d_in[2],  (const float*)d_in[3],
      (const float*)d_in[4],  (const float*)d_in[5],  (const float*)d_in[6],
      (const float*)d_in[7],  (const float*)d_in[8],  (const float*)d_in[9],
      (const float*)d_in[10], (const float*)d_in[11], out);

  mamba2_kernel<<<256, BLOCK, lds2, stream>>>(
      x,
      (const float*)d_in[12], (const float*)d_in[13], (const float*)d_in[14],
      (const float*)d_in[15], (const float*)d_in[16], (const float*)d_in[17],
      (const float*)d_in[18], (const float*)d_in[19], (const float*)d_in[20],
      (const float*)d_in[21], (const float*)d_in[22], out);
}
